// SlotAttention_30700426232600
// MI455X (gfx1250) — matrix operand order repaired and checked by reading, not compiled, not run on hardware
//
#include <hip/hip_runtime.h>
#include <hip/hip_bf16.h>

// ---------------------------------------------------------------------------
// SlotAttention on MI455X (gfx1250, wave32, WMMA bf16 path)
// ---------------------------------------------------------------------------
static constexpr int B_ = 32, N_ = 4096, D_ = 256, K_ = 8, SD_ = 64, HD_ = 128;
static constexpr int NUM_ITERS_ = 3;
static constexpr float EPS_ = 1e-8f, LN_EPS_ = 1e-5f;
static constexpr float SCALE_ = 0.125f;     // SD^-0.5 = 64^-0.5
static constexpr int NT_TOK = 1024;         // tokens per attention block

// packed bf16 weight buffer layout (dwords): [0,8192) W_proj, [8192,10240) Wk,
// [10240,12288) Wv
static constexpr int WPACK_WP = 8192, WPACK_WK = 2048, WPACK_WV = 2048;
static constexpr int WPACK_TOT = WPACK_WP + WPACK_WK + WPACK_WV;

typedef __attribute__((ext_vector_type(16))) __bf16 v16bf;
typedef __attribute__((ext_vector_type(8)))  float  v8f;
typedef __attribute__((vector_size(16)))     int    v4i_t;   // async-LDS payload

union ABfrag { unsigned u[8]; v16bf v; };

#if defined(__has_builtin)
#if __has_builtin(__builtin_amdgcn_global_load_async_to_lds_b128) && \
    __has_builtin(__builtin_amdgcn_s_wait_asynccnt)
#define HAVE_ASYNC_LDS 1
#endif
#endif
#ifndef HAVE_ASYNC_LDS
#define HAVE_ASYNC_LDS 0
#endif

#define AS_G __attribute__((address_space(1)))
#define AS_L __attribute__((address_space(3)))

__device__ __forceinline__ unsigned short f2bf(float f) {
  unsigned u = __float_as_uint(f);
  unsigned r = u + 0x7FFFu + ((u >> 16) & 1u);   // round-to-nearest-even
  return (unsigned short)(r >> 16);
}
__device__ __forceinline__ unsigned packbf(float lo, float hi) {
  return (unsigned)f2bf(lo) | ((unsigned)f2bf(hi) << 16);
}
__device__ __forceinline__ float wave_sum32(float v) {
  #pragma unroll
  for (int m = 16; m > 0; m >>= 1) v += __shfl_xor(v, m, 32);
  return v;
}
__device__ __forceinline__ v8f wmma_bf16(const ABfrag& a, const ABfrag& b, v8f c) {
  return __builtin_amdgcn_wmma_f32_16x16x32_bf16(false, a.v, false, b.v,
                                                 (short)0, c, false, false);
}

// ---------------------------------------------------------------------------
// Kernel 0 (runs once per call): pack W_proj / Wk / Wv to bf16-pair dwords so
// the 2048 projection blocks don't re-convert them.
// ---------------------------------------------------------------------------
__global__ void k_pack_weights(const float* __restrict__ Wp,
                               const float* __restrict__ Wk,
                               const float* __restrict__ Wv,
                               unsigned* __restrict__ out)
{
  int i = blockIdx.x * 256 + threadIdx.x;
  if (i < WPACK_WP) {
    out[i] = packbf(Wp[2 * i], Wp[2 * i + 1]);
  } else if (i < WPACK_WP + WPACK_WK) {
    int j = i - WPACK_WP;
    out[i] = packbf(Wk[2 * j], Wk[2 * j + 1]);
  } else if (i < WPACK_TOT) {
    int j = i - WPACK_WP - WPACK_WK;
    out[i] = packbf(Wv[2 * j], Wv[2 * j + 1]);
  }
}

// ---------------------------------------------------------------------------
// Kernel 1: fused LN(D) -> proj(D->SD) -> LN(SD) -> k,v GEMMs, write k/v as
// bf16 in WMMA-B-operand-friendly tiled layouts.
//   k_ws dwords: [B][N/16][p=0..31][t=0..15]  entry = (k[t][2p], k[t][2p+1])
//   v_ws dwords: [B][N/32][tp=0..15][d=0..63] entry = (v[2tp][d], v[2tp+1][d])
// 64 tokens per block, 256 threads (8 waves).
// ---------------------------------------------------------------------------
__global__ void __launch_bounds__(256)
k_proj_kv(const float* __restrict__ inputs,
          const float* __restrict__ lng, const float* __restrict__ lnb,
          const unsigned* __restrict__ wpack, const float* __restrict__ b_proj,
          const float* __restrict__ ln_in_g, const float* __restrict__ ln_in_b,
          unsigned* __restrict__ k_ws, unsigned* __restrict__ v_ws)
{
  extern __shared__ char smem[];
  unsigned* A   = (unsigned*)(smem);            // [64][128] bf16-pair dwords (32KB)
  unsigned* Wp  = (unsigned*)(smem + 32768);    // [64 out][128] pairs          (32KB)
  float*    XL  = (float*)   (smem + 65536);    // [64][64] f32                 (16KB)
  unsigned* A2  = (unsigned*)(smem + 81920);    // [64][32] pairs               (8KB)
  unsigned* Wkv = (unsigned*)(smem + 90112);    // WkL[2048] ++ WvL[2048]       (16KB)
  unsigned* WkL = Wkv;
  unsigned* WvL = Wkv + 2048;
  float*    KL  = (float*)(smem);               // reuse A region after GEMM1   (16KB)
  float*    VL  = (float*)(smem + 16384);       //                              (16KB)

  const int tid = threadIdx.x;
  const int lane = tid & 31, wave = tid >> 5;
  const int nl = lane & 15, hh = lane >> 4;
  const int b    = blockIdx.x >> 6;     // 64 blocks per batch
  const int c0   = blockIdx.x & 63;
  const int tok0 = c0 * 64;

  // ---- phase 0: stage pre-packed bf16 weights into LDS --------------------
#if HAVE_ASYNC_LDS
  for (int i = tid; i < 2048; i += 256)   // W_proj: 8192 dwords
    __builtin_amdgcn_global_load_async_to_lds_b128(
        (AS_G v4i_t*)(wpack + i * 4), (AS_L v4i_t*)(Wp + i * 4), 0, 0);
  for (int i = tid; i < 1024; i += 256)   // Wk ++ Wv: 4096 dwords (contiguous)
    __builtin_amdgcn_global_load_async_to_lds_b128(
        (AS_G v4i_t*)(wpack + WPACK_WP + i * 4), (AS_L v4i_t*)(Wkv + i * 4), 0, 0);
#else
  for (int i = tid; i < WPACK_WP; i += 256) Wp[i] = wpack[i];
  for (int i = tid; i < WPACK_WK + WPACK_WV; i += 256)
    Wkv[i] = wpack[WPACK_WP + i];
#endif

  // ---- phase 1: LN over D=256 per token, stage bf16 A ---------------------
  for (int row = wave; row < 64; row += 8) {
    const float* xr = inputs + ((size_t)b * N_ + tok0 + row) * D_;
    if (row + 8 < 64)                       // prefetch next row for this wave
      __builtin_prefetch(xr + 8 * D_ + lane * 8, 0, 3);
    float x8[8];
    #pragma unroll
    for (int j = 0; j < 8; j++) x8[j] = xr[lane * 8 + j];
    float s = 0.f;
    #pragma unroll
    for (int j = 0; j < 8; j++) s += x8[j];
    s = wave_sum32(s);
    float mean = s * (1.0f / 256.0f);
    float s2 = 0.f;
    #pragma unroll
    for (int j = 0; j < 8; j++) { float d = x8[j] - mean; s2 += d * d; }
    s2 = wave_sum32(s2);
    float rstd = rsqrtf(s2 * (1.0f / 256.0f) + LN_EPS_);
    #pragma unroll
    for (int j = 0; j < 8; j += 2) {
      int dc = lane * 8 + j;
      float a0 = (x8[j]     - mean) * rstd * lng[dc]     + lnb[dc];
      float a1 = (x8[j + 1] - mean) * rstd * lng[dc + 1] + lnb[dc + 1];
      A[row * 128 + (dc >> 1)] = packbf(a0, a1);
    }
  }
#if HAVE_ASYNC_LDS
  __builtin_amdgcn_s_wait_asynccnt(0);
#endif
  __syncthreads();

  // ---- phase 2: GEMM1  x[64x64] = A[64x256] * W_proj^T + b_proj -----------
  for (int tile = wave; tile < 16; tile += 8) {
    int mt = tile >> 2, nt = tile & 3;
    v8f acc;
    float bp = b_proj[nt * 16 + nl];
    #pragma unroll
    for (int r = 0; r < 8; r++) acc[r] = bp;
    #pragma unroll
    for (int kk = 0; kk < 8; kk++) {
      ABfrag a, bm;
      int m = mt * 16 + nl;
      #pragma unroll
      for (int r = 0; r < 8; r++) {
        int K0 = kk * 32 + (hh << 4) + (r << 1);
        a.u[r] = A[m * 128 + (K0 >> 1)];
      }
      int nout = nt * 16 + nl;
      #pragma unroll
      for (int r = 0; r < 8; r++)
        bm.u[r] = Wp[nout * 128 + kk * 16 + hh * 8 + r];
      acc = wmma_bf16(a, bm, acc);
    }
    #pragma unroll
    for (int r = 0; r < 8; r++)
      XL[(mt * 16 + r + hh * 8) * 64 + nt * 16 + nl] = acc[r];
  }
  __syncthreads();

  // ---- phase 3: LN over SD=64 per token -> bf16 A2 ------------------------
  for (int row = wave; row < 64; row += 8) {
    float a0 = XL[row * 64 + lane];
    float a1 = XL[row * 64 + lane + 32];
    float mean = wave_sum32(a0 + a1) * (1.0f / 64.0f);
    float d0 = a0 - mean, d1 = a1 - mean;
    float var = wave_sum32(d0 * d0 + d1 * d1) * (1.0f / 64.0f);
    float rstd = rsqrtf(var + LN_EPS_);
    unsigned short* A2h = (unsigned short*)A2;
    A2h[row * 64 + lane]      = f2bf(d0 * rstd * ln_in_g[lane]      + ln_in_b[lane]);
    A2h[row * 64 + lane + 32] = f2bf(d1 * rstd * ln_in_g[lane + 32] + ln_in_b[lane + 32]);
  }
  __syncthreads();

  // ---- phase 4: GEMM2  k = A2 * Wk^T,  v = A2 * Wv^T ----------------------
  for (int tile = wave; tile < 32; tile += 8) {
    int which = tile >> 4;           // 0 = k, 1 = v
    int rem = tile & 15;
    int mt = rem >> 2, nt = rem & 3;
    const unsigned* WT = which ? WvL : WkL;
    float* OUT = which ? VL : KL;
    v8f acc;
    #pragma unroll
    for (int r = 0; r < 8; r++) acc[r] = 0.f;
    #pragma unroll
    for (int kk = 0; kk < 2; kk++) {
      ABfrag a, bm;
      int m = mt * 16 + nl;
      #pragma unroll
      for (int r = 0; r < 8; r++) {
        int K0 = kk * 32 + (hh << 4) + (r << 1);
        a.u[r] = A2[m * 32 + (K0 >> 1)];
      }
      int nout = nt * 16 + nl;
      #pragma unroll
      for (int r = 0; r < 8; r++)
        bm.u[r] = WT[nout * 32 + kk * 16 + hh * 8 + r];
      acc = wmma_bf16(a, bm, acc);
    }
    #pragma unroll
    for (int r = 0; r < 8; r++)
      OUT[(mt * 16 + r + hh * 8) * 64 + nt * 16 + nl] = acc[r];
  }
  __syncthreads();

  // ---- phase 5: pack & store k_ws / v_ws ----------------------------------
  for (int i = tid; i < 2048; i += 256) {          // k: 4 chunks x 32 p x 16 t
    int cc = i >> 9, p = (i >> 4) & 31, t = i & 15;
    float lo = KL[(cc * 16 + t) * 64 + 2 * p];
    float hi = KL[(cc * 16 + t) * 64 + 2 * p + 1];
    size_t gchunk = (size_t)b * (N_ / 16) + c0 * 4 + cc;
    k_ws[(gchunk * 32 + p) * 16 + t] = packbf(lo, hi);
  }
  for (int i = tid; i < 2048; i += 256) {          // v: 2 chunks x 16 tp x 64 d
    int cc = i >> 10, tp = (i >> 6) & 15, d = i & 63;
    float lo = VL[(cc * 32 + 2 * tp)     * 64 + d];
    float hi = VL[(cc * 32 + 2 * tp + 1) * 64 + d];
    size_t gchunk = (size_t)b * (N_ / 32) + c0 * 2 + cc;
    v_ws[(gchunk * 16 + tp) * 64 + d] = packbf(lo, hi);
  }
}

// ---------------------------------------------------------------------------
// Kernel 2: slots = mu + exp(log_sigma) * noise
// ---------------------------------------------------------------------------
__global__ void k_init_slots(const float* __restrict__ noise,
                             const float* __restrict__ mu,
                             const float* __restrict__ lsig,
                             float* __restrict__ slots)
{
  int i = blockIdx.x * 256 + threadIdx.x;
  if (i < B_ * K_ * SD_) {
    int d = i & 63;
    slots[i] = mu[d] + __expf(lsig[d]) * noise[i];
  }
}

// ---------------------------------------------------------------------------
// Kernel 3: q = LN(slots) @ Wq^T * SCALE ; zero U,S ; optionally copy slots out
// One block per batch, 64 threads.
// ---------------------------------------------------------------------------
__global__ void __launch_bounds__(64)
k_qprep(const float* __restrict__ slots, const float* __restrict__ Wq,
        const float* __restrict__ g, const float* __restrict__ bb,
        float* __restrict__ q_ws, float* __restrict__ U, float* __restrict__ S,
        float* __restrict__ slots_out)
{
  __shared__ float sn[K_][SD_];
  int b = blockIdx.x, tid = threadIdx.x;
  for (int i = tid; i < K_ * SD_; i += 64) U[b * K_ * SD_ + i] = 0.f;
  if (tid < K_) S[b * K_ + tid] = 0.f;
  if (slots_out)
    for (int i = tid; i < K_ * SD_; i += 64)
      slots_out[b * K_ * SD_ + i] = slots[b * K_ * SD_ + i];
  if (tid < 32) {
    for (int s = 0; s < K_; s++) {
      float a0 = slots[(b * K_ + s) * SD_ + tid];
      float a1 = slots[(b * K_ + s) * SD_ + tid + 32];
      float mean = wave_sum32(a0 + a1) * (1.0f / 64.0f);
      float d0 = a0 - mean, d1 = a1 - mean;
      float var = wave_sum32(d0 * d0 + d1 * d1) * (1.0f / 64.0f);
      float rstd = rsqrtf(var + LN_EPS_);
      sn[s][tid]      = d0 * rstd * g[tid]      + bb[tid];
      sn[s][tid + 32] = d1 * rstd * g[tid + 32] + bb[tid + 32];
    }
  }
  __syncthreads();
  for (int s = 0; s < K_; s++) {
    float acc = 0.f;
    const float* wr = Wq + tid * SD_;
    #pragma unroll 8
    for (int i = 0; i < SD_; i++) acc += sn[s][i] * wr[i];
    q_ws[(b * K_ + s) * SD_ + tid] = acc * SCALE_;
  }
}

// ---------------------------------------------------------------------------
// Kernel 4: streaming attention pass.
//   logits = q k^T (WMMA), softmax over slots per token, then either
//   accumulate U = attn @ v (WMMA) and S = sum_n attn (atomics), or
//   (final_pass) write attn to d_out.
// 256 threads; each wave owns 32-token chunks.
// ---------------------------------------------------------------------------
__global__ void __launch_bounds__(256)
k_attn(const unsigned* __restrict__ k_ws, const unsigned* __restrict__ v_ws,
       const float* __restrict__ q_ws,
       float* __restrict__ U, float* __restrict__ S,
       float* __restrict__ attn_out, int final_pass)
{
  __shared__ unsigned qlds[16 * 32];                // [16 slot][32 dpair] bf16
  __shared__ unsigned short alds[8][16 * 32];       // per-wave [16 slot][32 n]
  const int tid = threadIdx.x, lane = tid & 31, wave = tid >> 5;
  const int nl = lane & 15, hh = lane >> 4;
  const int b  = blockIdx.x / (N_ / NT_TOK);
  const int nb = blockIdx.x % (N_ / NT_TOK);
  const int tokbase = nb * NT_TOK;

  for (int i = tid; i < 16 * 32; i += 256) {
    int m = i >> 5, dp = i & 31;
    unsigned val = 0u;
    if (m < K_)
      val = packbf(q_ws[(b * K_ + m) * SD_ + 2 * dp],
                   q_ws[(b * K_ + m) * SD_ + 2 * dp + 1]);
    qlds[i] = val;
  }
  __syncthreads();

  ABfrag aq[2];                                     // q A-operand, loop-invariant
  #pragma unroll
  for (int ks = 0; ks < 2; ks++)
    #pragma unroll
    for (int r = 0; r < 8; r++) {
      int K0 = ks * 32 + (hh << 4) + (r << 1);
      aq[ks].u[r] = qlds[nl * 32 + (K0 >> 1)];
    }

  v8f uacc[4];
  #pragma unroll
  for (int dt = 0; dt < 4; dt++)
    #pragma unroll
    for (int r = 0; r < 8; r++) uacc[dt][r] = 0.f;
  float sacc[8];
  #pragma unroll
  for (int r = 0; r < 8; r++) sacc[r] = 0.f;

  const unsigned* arow = (const unsigned*)&alds[wave][0];

  for (int ch = wave; ch < NT_TOK / 32; ch += 8) {
    int tok0 = tokbase + ch * 32;
    float attn_r[2][8];
    #pragma unroll
    for (int t = 0; t < 2; t++) {
      v8f lac;
      #pragma unroll
      for (int r = 0; r < 8; r++) lac[r] = 0.f;
      size_t gchunk = (size_t)b * (N_ / 16) + (tok0 + t * 16) / 16;
      #pragma unroll
      for (int ks = 0; ks < 2; ks++) {
        ABfrag kb;
        #pragma unroll
        for (int r = 0; r < 8; r++)
          kb.u[r] = k_ws[(gchunk * 32 + (ks * 16 + hh * 8 + r)) * 16 + nl];
        lac = wmma_bf16(aq[ks], kb, lac);
      }
      // softmax over the 8 slots (regs); hh==1 lanes hold padding rows.
      float mx = lac[0];
      #pragma unroll
      for (int r = 1; r < 8; r++) mx = fmaxf(mx, lac[r]);
      float den = 0.f;
      #pragma unroll
      for (int r = 0; r < 8; r++) { attn_r[t][r] = __expf(lac[r] - mx); den += attn_r[t][r]; }
      float inv = 1.0f / den;
      #pragma unroll
      for (int r = 0; r < 8; r++) {
        attn_r[t][r] *= inv;
        float av = (hh == 0) ? attn_r[t][r] : 0.f;
        alds[wave][(r + hh * 8) * 32 + t * 16 + nl] = f2bf(av);
        sacc[r] += av;
      }
    }
    if (final_pass) {
      if (hh == 0) {
        #pragma unroll
        for (int t = 0; t < 2; t++)
          #pragma unroll
          for (int r = 0; r < 8; r++)
            attn_out[((size_t)(b * K_ + r)) * N_ + tok0 + t * 16 + nl] = attn_r[t][r];
      }
    } else {
      ABfrag ua;                                   // attn^T as A-operand
      #pragma unroll
      for (int r = 0; r < 8; r++) {
        int K0 = (hh << 4) + (r << 1);
        ua.u[r] = arow[(nl * 32 + K0) >> 1];
      }
      size_t vchunk = (size_t)b * (N_ / 32) + tok0 / 32;
      #pragma unroll
      for (int dt = 0; dt < 4; dt++) {
        ABfrag vb;
        #pragma unroll
        for (int r = 0; r < 8; r++)
          vb.u[r] = v_ws[(vchunk * 16 + hh * 8 + r) * 64 + dt * 16 + nl];
        uacc[dt] = wmma_bf16(ua, vb, uacc[dt]);
      }
    }
  }

  if (!final_pass) {
    #pragma unroll
    for (int r = 0; r < 8; r++) {
      float v = sacc[r];                           // hh==1 lanes contributed 0
      #pragma unroll
      for (int m = 8; m > 0; m >>= 1) v += __shfl_xor(v, m, 32);
      if (lane == 0) atomicAdd(&S[b * K_ + r], v);
    }
    if (hh == 0) {
      #pragma unroll
      for (int dt = 0; dt < 4; dt++)
        #pragma unroll
        for (int r = 0; r < 8; r++)
          atomicAdd(&U[(b * K_ + r) * SD_ + dt * 16 + nl], uacc[dt][r]);
    }
  }
}

// ---------------------------------------------------------------------------
// Kernel 5: updates = U/(S+eps); GRU cell; residual MLP with LN.
// One block per (b, slot), 64 threads.
// ---------------------------------------------------------------------------
__global__ void __launch_bounds__(64)
k_gru_mlp(const float* __restrict__ U, const float* __restrict__ S,
          const float* __restrict__ slots_in, float* __restrict__ slots_out,
          const float* __restrict__ W_ih, const float* __restrict__ W_hh,
          const float* __restrict__ b_ih, const float* __restrict__ b_hh,
          const float* __restrict__ W_mlp1, const float* __restrict__ b_mlp1,
          const float* __restrict__ W_mlp2, const float* __restrict__ b_mlp2,
          const float* __restrict__ g_mlp, const float* __restrict__ b_mlpln)
{
  __shared__ float u[64], sp[64], lnm[64], hbuf[128], redA[2], redB[2];
  const int bs = blockIdx.x;
  const int b = bs >> 3, s = bs & 7;
  const int j = threadIdx.x;
  const int row = (b * K_ + s) * SD_;
  float Sv = S[b * K_ + s] + EPS_;
  u[j]  = U[row + j] / Sv;
  sp[j] = slots_in[row + j];
  __syncthreads();
  float gi[3], gh[3];
  #pragma unroll
  for (int gx = 0; gx < 3; gx++) {
    const float* wi = W_ih + (gx * 64 + j) * 64;
    const float* wh = W_hh + (gx * 64 + j) * 64;
    float a0 = 0.f, a1 = 0.f;
    #pragma unroll 8
    for (int i = 0; i < 64; i++) { a0 += u[i] * wi[i]; a1 += sp[i] * wh[i]; }
    gi[gx] = a0 + b_ih[gx * 64 + j];
    gh[gx] = a1 + b_hh[gx * 64 + j];
  }
  float r  = 1.0f / (1.0f + __expf(-(gi[0] + gh[0])));
  float z  = 1.0f / (1.0f + __expf(-(gi[1] + gh[1])));
  float nn = tanhf(gi[2] + r * gh[2]);
  float sm = (1.0f - z) * nn + z * sp[j];
  // LN over 64 across 2 waves
  float part = wave_sum32(sm);
  if ((j & 31) == 0) redA[j >> 5] = part;
  __syncthreads();
  float mean = (redA[0] + redA[1]) * (1.0f / 64.0f);
  float d = sm - mean;
  float p2 = wave_sum32(d * d);
  if ((j & 31) == 0) redB[j >> 5] = p2;
  __syncthreads();
  float var = (redB[0] + redB[1]) * (1.0f / 64.0f);
  float rstd = rsqrtf(var + LN_EPS_);
  lnm[j] = d * rstd * g_mlp[j] + b_mlpln[j];
  __syncthreads();
  #pragma unroll
  for (int t = 0; t < 2; t++) {
    const float* w1 = W_mlp1 + (t * 64 + j) * 64;
    float a = 0.f;
    #pragma unroll 8
    for (int i = 0; i < 64; i++) a += lnm[i] * w1[i];
    hbuf[t * 64 + j] = fmaxf(a + b_mlp1[t * 64 + j], 0.f);
  }
  __syncthreads();
  const float* w2 = W_mlp2 + j * 128;
  float a = 0.f;
  #pragma unroll 8
  for (int i = 0; i < 128; i++) a += hbuf[i] * w2[i];
  slots_out[row + j] = sm + a + b_mlp2[j];
}

// ---------------------------------------------------------------------------
extern "C" void kernel_launch(void* const* d_in, const int* in_sizes, int n_in,
                              void* d_out, int out_size, void* d_ws, size_t ws_size,
                              hipStream_t stream)
{
  (void)in_sizes; (void)n_in; (void)out_size; (void)ws_size;
  const float* inputs          = (const float*)d_in[0];
  const float* slot_noise      = (const float*)d_in[1];
  const float* slots_mu        = (const float*)d_in[2];
  const float* slots_log_sigma = (const float*)d_in[3];
  const float* ln_proj_g       = (const float*)d_in[4];
  const float* ln_proj_b       = (const float*)d_in[5];
  const float* W_proj          = (const float*)d_in[6];
  const float* b_proj          = (const float*)d_in[7];
  const float* Wq              = (const float*)d_in[8];
  const float* Wk              = (const float*)d_in[9];
  const float* Wv              = (const float*)d_in[10];
  const float* W_ih            = (const float*)d_in[11];
  const float* W_hh            = (const float*)d_in[12];
  const float* b_ih            = (const float*)d_in[13];
  const float* b_hh            = (const float*)d_in[14];
  const float* W_mlp1          = (const float*)d_in[15];
  const float* b_mlp1          = (const float*)d_in[16];
  const float* W_mlp2          = (const float*)d_in[17];
  const float* b_mlp2          = (const float*)d_in[18];
  const float* ln_in_g         = (const float*)d_in[19];
  const float* ln_in_b         = (const float*)d_in[20];
  const float* ln_slots_g      = (const float*)d_in[21];
  const float* ln_slots_b      = (const float*)d_in[22];
  const float* ln_mlp_g        = (const float*)d_in[23];
  const float* ln_mlp_b        = (const float*)d_in[24];

  char* ws = (char*)d_ws;
  size_t off = 0;
  auto take = [&](size_t bytes) -> char* {
    char* p = ws + off;
    off = (off + bytes + 255) & ~(size_t)255;
    return p;
  };
  unsigned* k_ws  = (unsigned*)take((size_t)B_ * N_ * SD_ * 2);   // bf16
  unsigned* v_ws  = (unsigned*)take((size_t)B_ * N_ * SD_ * 2);   // bf16
  unsigned* wpack = (unsigned*)take((size_t)WPACK_TOT * 4);
  float*    q_ws  = (float*)take((size_t)B_ * K_ * SD_ * 4);
  float*    U     = (float*)take((size_t)B_ * K_ * SD_ * 4);
  float*    S     = (float*)take((size_t)B_ * K_ * 4);
  float*    slots = (float*)take((size_t)B_ * K_ * SD_ * 4);

  float* out       = (float*)d_out;
  float* slots_out = out;                       // [B,K,SD]
  float* attn_out  = out + B_ * K_ * SD_;       // [B,K,N]

  k_pack_weights<<<(WPACK_TOT + 255) / 256, 256, 0, stream>>>(W_proj, Wk, Wv,
                                                              wpack);
  k_proj_kv<<<B_ * (N_ / 64), 256, 106496, stream>>>(
      inputs, ln_proj_g, ln_proj_b, wpack, b_proj,
      ln_in_g, ln_in_b, k_ws, v_ws);
  k_init_slots<<<(B_ * K_ * SD_ + 255) / 256, 256, 0, stream>>>(
      slot_noise, slots_mu, slots_log_sigma, slots);

  for (int it = 0; it < NUM_ITERS_; it++) {
    k_qprep<<<B_, 64, 0, stream>>>(slots, Wq, ln_slots_g, ln_slots_b,
                                   q_ws, U, S, nullptr);
    k_attn<<<B_ * (N_ / NT_TOK), 256, 0, stream>>>(k_ws, v_ws, q_ws, U, S,
                                                   nullptr, 0);
    k_gru_mlp<<<B_ * K_, 64, 0, stream>>>(U, S, slots, slots,
                                          W_ih, W_hh, b_ih, b_hh,
                                          W_mlp1, b_mlp1, W_mlp2, b_mlp2,
                                          ln_mlp_g, ln_mlp_b);
  }
  k_qprep<<<B_, 64, 0, stream>>>(slots, Wq, ln_slots_g, ln_slots_b,
                                 q_ws, U, S, slots_out);
  k_attn<<<B_ * (N_ / NT_TOK), 256, 0, stream>>>(k_ws, v_ws, q_ws, U, S,
                                                 attn_out, 1);
}
